// LSTM_54417235641056
// MI455X (gfx1250) — matrix operand order repaired and checked by reading, not compiled, not run on hardware
//
#include <hip/hip_runtime.h>

typedef __attribute__((ext_vector_type(16))) _Float16 v16h;
typedef __attribute__((ext_vector_type(8)))  _Float16 v8h;
typedef __attribute__((ext_vector_type(8)))  float    v8f;

// Problem dims
constexpr int T  = 512;   // sequence length
constexpr int BB = 64;    // batch
constexpr int I  = 256;   // input size
constexpr int H  = 512;   // hidden size
constexpr int NG = 4;     // gates (f, i, o, c)
constexpr int KT = (I + H) / 32;   // 24 k-ticks of 32 (8 for x, 16 for h)
constexpr int NBLK  = H / 16;      // 32 blocks, one 16-unit tile each
constexpr int NWAVE = BB / 16;     // 4 waves/block, one 16-batch tile each

constexpr int WT_HALVES = NBLK * NG * KT * 512;  // 1,572,864 packed f16 weights

// Workspace layout (bytes), all regions 256B aligned by construction
constexpr size_t OFF_W    = 0;
constexpr size_t SZ_W     = (size_t)WT_HALVES * 2;        // 3,145,728
constexpr size_t OFF_X    = OFF_W + SZ_W;
constexpr size_t SZ_X     = (size_t)T * BB * I * 2;       // 16,777,216
constexpr size_t OFF_H    = OFF_X + SZ_X;
constexpr size_t SZ_H     = (size_t)2 * BB * H * 2;       // 131,072 (double buffer)
constexpr size_t OFF_BIAS = OFF_H + SZ_H;
constexpr size_t SZ_BIAS  = (size_t)NG * H * 4;           // 8,192
constexpr size_t OFF_CNT  = OFF_BIAS + SZ_BIAS;           // 512 * 4 counters

__device__ __forceinline__ float fast_tanh(float x) {
#if __has_builtin(__builtin_amdgcn_tanhf)
  return __builtin_amdgcn_tanhf(x);     // v_tanh_f32 on gfx1250
#else
  return tanhf(x);
#endif
}
__device__ __forceinline__ float fast_sigmoid(float x) {
  return 0.5f + 0.5f * fast_tanh(0.5f * x);
}

__device__ __forceinline__ v16h cat16(v8h lo, v8h hi) {
  return __builtin_shufflevector(lo, hi, 0,1,2,3,4,5,6,7,8,9,10,11,12,13,14,15);
}

__device__ __forceinline__ v8f wmma16(v16h a, v16h b, v8f c) {
  // D = A(16x32 f16) * B(32x16 f16) + C(16x16 f32)
  return __builtin_amdgcn_wmma_f32_16x16x32_f16(false, a, false, b, (short)0, c,
                                                false, false);
}

// ---------------------------------------------------------------------------
// Prep 1: pack [Wx | Wh] (fp32) into f16 B-matrix register-image tiles and
// build the combined bias.  Tile (ut, g, kt) holds B[k][n] for
// n = g*512 + ut*16 + (lane&15), k = kt*32 + (j>>3)*16 + (lane>>4)*8 + (j&7)  (j = 0..15).
// Stored lane-major: halves[ (((ut*4+g)*24+kt)*32 + lane)*16 + j ].
// ---------------------------------------------------------------------------
__global__ void lstm_prep_weights(const float* __restrict__ Wx,
                                  const float* __restrict__ Wh,
                                  const float* __restrict__ bx,
                                  const float* __restrict__ bh,
                                  _Float16* __restrict__ wt,
                                  float* __restrict__ bias) {
  int id = blockIdx.x * blockDim.x + threadIdx.x;
  if (id < WT_HALVES) {
    int j    = id & 15;
    int lane = (id >> 4) & 31;
    int rest = id >> 9;             // (ut*4 + g)*24 + kt
    int kt   = rest % KT;
    rest    /= KT;                  // ut*4 + g
    int g    = rest & 3;
    int ut   = rest >> 2;
    int hi   = lane >> 4;
    int u    = ut * 16 + (lane & 15);
    int k    = kt * 32 + (j >> 3) * 16 + hi * 8 + (j & 7);     // B-matrix k index for element j
    float v  = (k < I) ? Wx[((size_t)g * H + u) * I + k]
                       : Wh[((size_t)g * H + u) * H + (k - I)];
    wt[id] = (_Float16)v;
  }
  if (id < NG * H) bias[id] = bx[id] + bh[id];
}

// ---------------------------------------------------------------------------
// Prep 2: x fp32 -> f16 (row-major [t][b][i]), zero the h double buffer and
// the per-step barrier counters.  Re-run every launch (determinism).
// ---------------------------------------------------------------------------
__global__ void lstm_prep_x(const float* __restrict__ x,
                            _Float16* __restrict__ xf,
                            _Float16* __restrict__ hbuf,
                            unsigned* __restrict__ cnt) {
  int id = blockIdx.x * blockDim.x + threadIdx.x;
  if (id < T * BB * I) xf[id] = (_Float16)x[id];
  if (id < 2 * BB * H) hbuf[id] = (_Float16)0.f;
  if (id < T) cnt[id] = 0u;
}

// ---------------------------------------------------------------------------
// Persistent LSTM: 32 blocks x 128 threads.  Block = one 16-unit column tile
// (all 4 gates in its 96KB LDS slice); wave = one 16-batch row tile.
// Per step: 24 k-ticks x 4 gates of v_wmma_f32_16x16x32_f16, then the cell
// update entirely in registers, then a grid-wide step barrier.
// ---------------------------------------------------------------------------
__global__ __launch_bounds__(NWAVE * 32, 1)
void lstm_persistent(const _Float16* __restrict__ xf,
                     const _Float16* __restrict__ wt,
                     const float* __restrict__ bias,
                     _Float16* hbuf,
                     unsigned* cnt,
                     float* __restrict__ out) {
  __shared__ _Float16 wlds[NG * KT * 512];   // 96 KB of weight tiles

  const int tid  = threadIdx.x;
  const int lane = tid & 31;
  const int wave = tid >> 5;      // batch tile
  const int ut   = blockIdx.x;    // unit tile

  // Stage this block's weight tiles into LDS (contiguous 16B chunks).
  {
    const uint4* src = (const uint4*)(wt + (size_t)ut * (NG * KT * 512));
    uint4* dst = (uint4*)wlds;
    for (int i = tid; i < (NG * KT * 512 * 2) / 16; i += NWAVE * 32) dst[i] = src[i];
  }
  __syncthreads();

  const int hi   = lane >> 4;
  const int ml   = lane & 15;
  const int bRow = wave * 16 + ml;        // batch row fed into A by this lane
  const int uCol = ut * 16 + ml;          // unit column owned by this lane
  const int bOut = wave * 16 + hi * 8;    // first output batch row (C/D layout)

  float bcol[NG];
#pragma unroll
  for (int g = 0; g < NG; ++g) bcol[g] = bias[g * H + uCol];

  v8f c = {};                              // persistent cell state

  for (int t = 0; t < T; ++t) {
    v8f accE[NG] = {};
    v8f accO[NG] = {};

    const _Float16* xrow = xf + ((size_t)t * BB + bRow) * I + hi * 8;
    const _Float16* hrow =
        hbuf + (size_t)(t & 1) * (BB * H) + (size_t)bRow * H + hi * 8;

#pragma unroll
    for (int kt = 0; kt < KT; ++kt) {
      // A tile: rows = batch, k in [kt*32, kt*32+32); x for kt<8 else h_prev.
      const _Float16* ap = (kt < 8) ? (xrow + kt * 32) : (hrow + (kt - 8) * 32);
      v8h alo = *(const v8h*)ap;           // k = hi*8 + 0..7
      v8h ahi = *(const v8h*)(ap + 16);    // k = 16 + hi*8 + 0..7
      v16h A = cat16(alo, ahi);
#pragma unroll
      for (int g = 0; g < NG; ++g) {
        const _Float16* bp = &wlds[((g * KT + kt) * 32 + lane) * 16];
        v8h blo = *(const v8h*)bp;
        v8h bhi = *(const v8h*)(bp + 8);
        v16h Bm = cat16(blo, bhi);
        if (kt & 1) accO[g] = wmma16(A, Bm, accO[g]);
        else        accE[g] = wmma16(A, Bm, accE[g]);
      }
    }

    // Warm L2/WGP$ for next timestep's x slice.
    if (t + 1 < T)
      __builtin_prefetch(xf + ((size_t)(t + 1) * BB + bRow) * I, 0, 0);

    _Float16* hnext = hbuf + (size_t)((t + 1) & 1) * (BB * H);
#pragma unroll
    for (int j = 0; j < 8; ++j) {
      float pf = accE[0][j] + accO[0][j] + bcol[0];
      float pi = accE[1][j] + accO[1][j] + bcol[1];
      float po = accE[2][j] + accO[2][j] + bcol[2];
      float pc = accE[3][j] + accO[3][j] + bcol[3];
      float f  = fast_sigmoid(pf);
      float i  = fast_sigmoid(pi);
      float o  = fast_sigmoid(po);
      float cn = f * c[j] + i * fast_tanh(pc);
      c[j] = cn;
      float h = o * fast_tanh(cn);
      int b = bOut + j;
      hnext[(size_t)b * H + uCol] = (_Float16)h;             // f16 state
      out[((size_t)t * BB + b) * H + uCol] = h;              // h_seq
      if (t == T - 1) {
        out[(size_t)T * BB * H + (size_t)b * H + uCol] = h;                    // h_last
        out[(size_t)T * BB * H + (size_t)BB * H + (size_t)b * H + uCol] = cn;  // c_last
      }
    }

    // Grid-wide step barrier (step-indexed counters: no reset races).
    if (t + 1 < T) {
      __threadfence();        // release: make h stores device-visible
      __syncthreads();
      if (tid == 0) {
        atomicAdd(&cnt[t], 1u);
        while (atomicAdd(&cnt[t], 0u) < (unsigned)gridDim.x)
          __builtin_amdgcn_s_sleep(1);
      }
      __syncthreads();
      __threadfence();        // acquire: invalidate WGP$ before reading h
    }
  }
}

// ---------------------------------------------------------------------------
// Inputs (setup_inputs order): x[T,B,I] f32, Wx[4,H,I] f32, bx[4,H] f32,
// Wh[4,H,H] f32, bh[4,H] f32.  Output: h_seq[T,B,H] ++ h_last[B,H] ++
// c_last[B,H], all f32.
// ---------------------------------------------------------------------------
extern "C" void kernel_launch(void* const* d_in, const int* in_sizes, int n_in,
                              void* d_out, int out_size, void* d_ws,
                              size_t ws_size, hipStream_t stream) {
  const float* x  = (const float*)d_in[0];
  const float* Wx = (const float*)d_in[1];
  const float* bx = (const float*)d_in[2];
  const float* Wh = (const float*)d_in[3];
  const float* bh = (const float*)d_in[4];
  float* out = (float*)d_out;

  char* ws = (char*)d_ws;
  _Float16* wt   = (_Float16*)(ws + OFF_W);
  _Float16* xf   = (_Float16*)(ws + OFF_X);
  _Float16* hbuf = (_Float16*)(ws + OFF_H);
  float*    bias = (float*)(ws + OFF_BIAS);
  unsigned* cnt  = (unsigned*)(ws + OFF_CNT);

  lstm_prep_weights<<<WT_HALVES / 256, 256, 0, stream>>>(Wx, Wh, bx, bh, wt, bias);
  lstm_prep_x<<<(T * BB * I) / 256, 256, 0, stream>>>(x, xf, hbuf, cnt);
  lstm_persistent<<<NBLK, NWAVE * 32, 0, stream>>>(xf, wt, bias, hbuf, cnt, out);
}
